// AttentionRecursive_31464930410675
// MI455X (gfx1250) — matrix operand order repaired and checked by reading, not compiled, not run on hardware
//
#include <hip/hip_runtime.h>
#include <hip/hip_bf16.h>

#define DEVINL __device__ __forceinline__

typedef __attribute__((ext_vector_type(16))) __bf16 v16bf;
typedef __attribute__((ext_vector_type(8)))  float  v8f;

// Problem constants
#define BB    128
#define TT    512
#define FF    32
#define HH    512
#define AA    512
#define HORZ  24
#define G4    2048   // 4*H

DEVINL float sigmoidf_(float x) { return 1.0f / (1.0f + __expf(-x)); }

// ------------------------------------------------------------------
// WMMA bf16 fragment loaders (CDNA5 ISA 7.12.2 VGPR layouts, wave32)
// ------------------------------------------------------------------
// A: row-major [M x K], 16x32 tile at (row0,k0), row stride lda.
// lanes 0-15 : row=lane,    K = {k0..k0+7} u {k0+16..k0+23}
// lanes 16-31: row=lane-16, K = {k0+8..k0+15} u {k0+24..k0+31}
DEVINL v16bf load_a_frag(const __bf16* A, int lda, int row0, int k0, int lane) {
  const int r  = lane & 15;
  const int hi = lane >> 4;
  const __bf16* p = A + (size_t)(row0 + r) * lda + k0 + hi * 8;
  v16bf a;
#pragma unroll
  for (int j = 0; j < 8; ++j) { a[j] = p[j]; a[8 + j] = p[16 + j]; }
  return a;
}

// B: from pre-transposed weight Wt row-major [K x N]; 32x16 tile at (k0,n0).
// lane l holds B column n0+(l&15), K rows as in load_a_frag.
DEVINL v16bf load_b_frag(const __bf16* Wt, int ldb, int k0, int n0, int lane) {
  const __bf16* p = Wt + (size_t)(k0 + (lane >> 4) * 8) * ldb + n0 + (lane & 15);
  v16bf b;
#pragma unroll
  for (int j = 0; j < 8; ++j) { b[j] = p[(size_t)j * ldb]; b[8 + j] = p[(size_t)(16 + j) * ldb]; }
  return b;
}

DEVINL v8f wmma_bf16(v16bf a, v16bf b, v8f c) {
  return __builtin_amdgcn_wmma_f32_16x16x32_bf16(false, a, false, b,
                                                 (short)0, c, false, false);
}

// ------------------------------------------------------------------
// Prep kernels
// ------------------------------------------------------------------
// dst[R x C] bf16 = transpose of src[C x R] f32
__global__ void k_transpose_bf16(const float* __restrict__ src,
                                 __bf16* __restrict__ dst, int R, int C) {
  size_t n = (size_t)R * C;
  for (size_t i = (size_t)blockIdx.x * blockDim.x + threadIdx.x; i < n;
       i += (size_t)gridDim.x * blockDim.x) {
    size_t r = i / C, c = i % C;
    dst[i] = (__bf16)src[c * (size_t)R + r];
  }
}

// x [B,T,F] f32 -> xbf [T,B,F] bf16
__global__ void k_x_to_bf(const float* __restrict__ x, __bf16* __restrict__ xbf) {
  size_t n = (size_t)TT * BB * FF;
  for (size_t i = (size_t)blockIdx.x * blockDim.x + threadIdx.x; i < n;
       i += (size_t)gridDim.x * blockDim.x) {
    size_t t = i / (BB * FF);
    size_t rem = i % (BB * FF);
    size_t b = rem / FF, f = rem % FF;
    xbf[i] = (__bf16)x[(b * TT + t) * FF + f];
  }
}

__global__ void k_bias_sum(const float* __restrict__ a, const float* __restrict__ b,
                           float* __restrict__ o, int n) {
  for (int i = blockIdx.x * blockDim.x + threadIdx.x; i < n;
       i += gridDim.x * blockDim.x)
    o[i] = a[i] + b[i];
}

// ------------------------------------------------------------------
// Fused LSTM step: gates = Xt@Wxt + Hprev@Whht + bsum; act; update h,c
// grid (H/16, B/16), 128 threads = 4 waves, wave w computes gate w tile.
// ------------------------------------------------------------------
__global__ __launch_bounds__(128)
void k_lstm_step(const __bf16* __restrict__ Xt, int Kx,
                 const __bf16* __restrict__ Wxt,
                 const __bf16* __restrict__ Hprev,
                 const __bf16* __restrict__ Whht,
                 const float* __restrict__ bsum,
                 const float* __restrict__ Cprev,
                 float* __restrict__ Cnew,
                 __bf16* __restrict__ Hnew_bf,
                 __bf16* __restrict__ Seq_bf) {
  __shared__ float lds_g[4][16][16];
  const int tid  = threadIdx.x;
  const int wave = tid >> 5;        // gate id (i,f,g,o)
  const int lane = tid & 31;
  const int ntile = blockIdx.x;     // 0..31
  const int rtile = blockIdx.y;     // 0..7
  const int row0  = rtile * 16;
  const int gcol0 = wave * HH + ntile * 16;

  v8f acc = {};
  for (int k = 0; k < Kx; k += 32) {
    v16bf a = load_a_frag(Xt, Kx, row0, k, lane);
    v16bf b = load_b_frag(Wxt, G4, k, gcol0, lane);
    acc = wmma_bf16(a, b, acc);
  }
  for (int k = 0; k < HH; k += 32) {
    v16bf a = load_a_frag(Hprev, HH, row0, k, lane);
    v16bf b = load_b_frag(Whht, G4, k, gcol0, lane);
    acc = wmma_bf16(a, b, acc);
  }
  const int nl = lane & 15, hi = lane >> 4;
  const float bias = bsum[gcol0 + nl];
#pragma unroll
  for (int i = 0; i < 8; ++i)
    lds_g[wave][hi * 8 + i][nl] = acc[i] + bias;
  __syncthreads();

#pragma unroll
  for (int e = 0; e < 2; ++e) {
    int idx = tid + e * 128;
    int m = idx >> 4, n = idx & 15;
    int brow = row0 + m;
    int hcol = ntile * 16 + n;
    float gi = lds_g[0][m][n], gf = lds_g[1][m][n];
    float gg = lds_g[2][m][n], go = lds_g[3][m][n];
    float c  = Cprev[(size_t)brow * HH + hcol];
    float cn = sigmoidf_(gf) * c + sigmoidf_(gi) * tanhf(gg);
    float hn = sigmoidf_(go) * tanhf(cn);
    Cnew[(size_t)brow * HH + hcol] = cn;
    __bf16 hb = (__bf16)hn;
    Hnew_bf[(size_t)brow * HH + hcol] = hb;
    Seq_bf[(size_t)brow * HH + hcol]  = hb;
  }
}

// ------------------------------------------------------------------
// Encoder attention projection (done ONCE): PE = lstm_out @ Wattn_enc^T + battn
// M = T*B = 65536, N = A, K = H.  8 waves/WG, one 16x16 tile per wave.
// ------------------------------------------------------------------
__global__ __launch_bounds__(256)
void k_encproj(const __bf16* __restrict__ LO,   // [T*B, H]
               const __bf16* __restrict__ WaT,  // [2H, A], rows 0..H-1 used
               const float* __restrict__ battn,
               __bf16* __restrict__ PE) {       // [T*B, A]
  const int wave = threadIdx.x >> 5, lane = threadIdx.x & 31;
  const long tile = (long)blockIdx.x * 8 + wave;
  const int  nt = (int)(tile & 31);
  const long mt = tile >> 5;
  const int  row0 = (int)(mt * 16), col0 = nt * 16;
  v8f acc = {};
  for (int k = 0; k < HH; k += 32) {
    v16bf a = load_a_frag(LO, HH, row0, k, lane);
    v16bf b = load_b_frag(WaT, AA, k, col0, lane);
    acc = wmma_bf16(a, b, acc);
  }
  const int nl = lane & 15, hi = lane >> 4;
  const float bias = battn[col0 + nl];
#pragma unroll
  for (int i = 0; i < 8; ++i)
    PE[(size_t)(row0 + hi * 8 + i) * AA + col0 + nl] = (__bf16)(acc[i] + bias);
}

// Per-decoder-step: hW = h @ Wattn_dec^T   (M=B, N=A, K=H). 32 WGs of 8 waves.
__global__ __launch_bounds__(256)
void k_dechproj(const __bf16* __restrict__ Hbf,  // [B, H]
                const __bf16* __restrict__ WaTd, // [H, A] (dec half)
                float* __restrict__ hW) {        // [B, A]
  const int wave = threadIdx.x >> 5, lane = threadIdx.x & 31;
  const int tile = blockIdx.x * 8 + wave;
  const int nt = tile & 31, mt = tile >> 5;
  const int row0 = mt * 16, col0 = nt * 16;
  v8f acc = {};
  for (int k = 0; k < HH; k += 32) {
    v16bf a = load_a_frag(Hbf, HH, row0, k, lane);
    v16bf b = load_b_frag(WaTd, AA, k, col0, lane);
    acc = wmma_bf16(a, b, acc);
  }
  const int nl = lane & 15, hi = lane >> 4;
#pragma unroll
  for (int i = 0; i < 8; ++i)
    hW[(size_t)(row0 + hi * 8 + i) * AA + col0 + nl] = acc[i];
}

// ------------------------------------------------------------------
// Attention scores + softmax + context, one block per batch element.
// scores[t] = sum_a v[a]*tanh(PE[t,b,a] + hW[b,a]); softmax over t;
// ctx[b,:] = sum_t w[t]*LO[t,b,:]
// ------------------------------------------------------------------
__global__ __launch_bounds__(256)
void k_attn(const __bf16* __restrict__ PE, const float* __restrict__ hW,
            const float* __restrict__ vvec, const __bf16* __restrict__ LO,
            float* __restrict__ ctx) {
  __shared__ float s_scores[TT];
  __shared__ float s_hw[AA];
  __shared__ float s_v[AA];
  __shared__ float s_red[8];
  const int b = blockIdx.x, tid = threadIdx.x;
  const int wave = tid >> 5, lane = tid & 31;

  s_hw[tid] = hW[(size_t)b * AA + tid];
  s_hw[tid + 256] = hW[(size_t)b * AA + tid + 256];
  s_v[tid] = vvec[tid];
  s_v[tid + 256] = vvec[tid + 256];
  __syncthreads();

  for (int t = wave; t < TT; t += 8) {
    const __bf16* pe = PE + ((size_t)t * BB + b) * AA;
    float s = 0.0f;
    for (int a = lane; a < AA; a += 32)
      s += s_v[a] * tanhf((float)pe[a] + s_hw[a]);
#pragma unroll
    for (int off = 16; off; off >>= 1) s += __shfl_xor(s, off, 32);
    if (lane == 0) s_scores[t] = s;
  }
  __syncthreads();

  // softmax (unnormalized exp kept in s_scores; scale factor folded into ctx)
  float mx = -1e30f;
  for (int i = tid; i < TT; i += 256) mx = fmaxf(mx, s_scores[i]);
#pragma unroll
  for (int off = 16; off; off >>= 1) mx = fmaxf(mx, __shfl_xor(mx, off, 32));
  if (lane == 0) s_red[wave] = mx;
  __syncthreads();
  mx = s_red[0];
#pragma unroll
  for (int i = 1; i < 8; ++i) mx = fmaxf(mx, s_red[i]);
  __syncthreads();                       // protect s_red reuse

  float sum = 0.0f;
  for (int i = tid; i < TT; i += 256) {
    float e = __expf(s_scores[i] - mx);
    s_scores[i] = e;
    sum += e;
  }
#pragma unroll
  for (int off = 16; off; off >>= 1) sum += __shfl_xor(sum, off, 32);
  if (lane == 0) s_red[wave] = sum;
  __syncthreads();
  float S = 0.0f;
#pragma unroll
  for (int i = 0; i < 8; ++i) S += s_red[i];
  const float invS = 1.0f / S;

  for (int h = tid; h < HH; h += 256) {
    float acc = 0.0f;
    for (int t = 0; t < TT; ++t)
      acc += s_scores[t] * (float)LO[((size_t)t * BB + b) * HH + h];
    ctx[(size_t)b * HH + h] = acc * invS;
  }
}

// ------------------------------------------------------------------
// Decoder LSTMCell step: gates = xt*WihC + h@WhhCt + bsum (rank-1 input)
// ------------------------------------------------------------------
__global__ __launch_bounds__(128)
void k_dec_cell(const float* __restrict__ yy, int step,
                const float* __restrict__ WihC,   // [4H]
                const __bf16* __restrict__ Hprev, // [B,H] bf16
                const __bf16* __restrict__ WhhCt, // [H,4H]
                const float* __restrict__ bsum,
                const float* __restrict__ Cprev,
                float* __restrict__ Cnew,
                __bf16* __restrict__ Hnew_bf,
                float* __restrict__ Hnew_f32) {
  __shared__ float lds_g[4][16][16];
  const int tid = threadIdx.x, wave = tid >> 5, lane = tid & 31;
  const int ntile = blockIdx.x, rtile = blockIdx.y;
  const int row0 = rtile * 16;
  const int gcol0 = wave * HH + ntile * 16;

  v8f acc = {};
  for (int k = 0; k < HH; k += 32) {
    v16bf a = load_a_frag(Hprev, HH, row0, k, lane);
    v16bf b = load_b_frag(WhhCt, G4, k, gcol0, lane);
    acc = wmma_bf16(a, b, acc);
  }
  const int nl = lane & 15, hi = lane >> 4;
  const float bias = bsum[gcol0 + nl];
#pragma unroll
  for (int i = 0; i < 8; ++i)
    lds_g[wave][hi * 8 + i][nl] = acc[i] + bias;
  __syncthreads();

#pragma unroll
  for (int e = 0; e < 2; ++e) {
    int idx = tid + e * 128;
    int m = idx >> 4, n = idx & 15;
    int brow = row0 + m;
    int hcol = ntile * 16 + n;
    float xb = (step == 0) ? 0.0f : yy[(size_t)brow * HORZ + (step - 1)];
    float gi = lds_g[0][m][n] + xb * WihC[0 * HH + hcol];
    float gf = lds_g[1][m][n] + xb * WihC[1 * HH + hcol];
    float gg = lds_g[2][m][n] + xb * WihC[2 * HH + hcol];
    float go = lds_g[3][m][n] + xb * WihC[3 * HH + hcol];
    float c  = Cprev[(size_t)brow * HH + hcol];
    float cn = sigmoidf_(gf) * c + sigmoidf_(gi) * tanhf(gg);
    float hn = sigmoidf_(go) * tanhf(cn);
    Cnew[(size_t)brow * HH + hcol] = cn;
    Hnew_bf[(size_t)brow * HH + hcol] = (__bf16)hn;
    Hnew_f32[(size_t)brow * HH + hcol] = hn;
  }
}

// forecast: out[b,step] = Wf[:H].h_new[b] + Wf[H:].ctx[b] + bf
__global__ __launch_bounds__(256)
void k_forecast(const float* __restrict__ Hnew, const float* __restrict__ ctx,
                const float* __restrict__ Wf, const float* __restrict__ bfp,
                float* __restrict__ out, int step) {
  __shared__ float red[8];
  const int b = blockIdx.x, tid = threadIdx.x;
  float s = 0.0f;
  for (int k = tid; k < 2 * HH; k += 256) {
    float xv = (k < HH) ? Hnew[(size_t)b * HH + k] : ctx[(size_t)b * HH + (k - HH)];
    s += xv * Wf[k];
  }
#pragma unroll
  for (int off = 16; off; off >>= 1) s += __shfl_xor(s, off, 32);
  if ((tid & 31) == 0) red[tid >> 5] = s;
  __syncthreads();
  if (tid == 0) {
    float t = 0.0f;
#pragma unroll
    for (int i = 0; i < 8; ++i) t += red[i];
    out[(size_t)b * HORZ + step] = t + bfp[0];
  }
}

// ------------------------------------------------------------------
extern "C" void kernel_launch(void* const* d_in, const int* in_sizes, int n_in,
                              void* d_out, int out_size, void* d_ws, size_t ws_size,
                              hipStream_t stream) {
  (void)in_sizes; (void)n_in; (void)out_size; (void)ws_size;
  const float* x     = (const float*)d_in[0];
  const float* y     = (const float*)d_in[1];
  const float* Wih0  = (const float*)d_in[2];
  const float* Whh0  = (const float*)d_in[3];
  const float* bih0  = (const float*)d_in[4];
  const float* bhh0  = (const float*)d_in[5];
  const float* Wih1  = (const float*)d_in[6];
  const float* Whh1  = (const float*)d_in[7];
  const float* bih1  = (const float*)d_in[8];
  const float* bhh1  = (const float*)d_in[9];
  const float* Wattn = (const float*)d_in[10];
  const float* battn = (const float*)d_in[11];
  const float* vvec  = (const float*)d_in[12];
  const float* WihC  = (const float*)d_in[13];
  const float* WhhC  = (const float*)d_in[14];
  const float* bihC  = (const float*)d_in[15];
  const float* bhhC  = (const float*)d_in[16];
  const float* Wf    = (const float*)d_in[17];
  const float* bfp   = (const float*)d_in[18];
  float* out = (float*)d_out;

  char* w = (char*)d_ws;
  size_t off = 0;
  auto carve = [&](size_t bytes) -> void* {
    void* p = w + off;
    off = (off + bytes + 255) & ~(size_t)255;
    return p;
  };

  __bf16* xbf    = (__bf16*)carve((size_t)TT * BB * FF * 2);
  __bf16* Wih0t  = (__bf16*)carve((size_t)FF * G4 * 2);
  __bf16* Whh0t  = (__bf16*)carve((size_t)HH * G4 * 2);
  __bf16* Wih1t  = (__bf16*)carve((size_t)HH * G4 * 2);
  __bf16* Whh1t  = (__bf16*)carve((size_t)HH * G4 * 2);
  __bf16* WhhCt  = (__bf16*)carve((size_t)HH * G4 * 2);
  __bf16* WaT    = (__bf16*)carve((size_t)2 * HH * AA * 2);  // [2H, A]
  float*  bsum0  = (float*)carve(G4 * 4);
  float*  bsum1  = (float*)carve(G4 * 4);
  float*  bsumC  = (float*)carve(G4 * 4);
  __bf16* out0bf = (__bf16*)carve((size_t)TT * BB * HH * 2); // [T,B,H]
  __bf16* lobf   = (__bf16*)carve((size_t)TT * BB * HH * 2); // [T,B,H]
  __bf16* pebf   = (__bf16*)carve((size_t)TT * BB * AA * 2); // [T,B,A]
  __bf16* henc0  = (__bf16*)carve((size_t)BB * HH * 2);
  __bf16* henc1  = (__bf16*)carve((size_t)BB * HH * 2);
  float*  cenc0  = (float*)carve((size_t)BB * HH * 4);
  float*  cenc1  = (float*)carve((size_t)BB * HH * 4);
  __bf16* hdecb0 = (__bf16*)carve((size_t)BB * HH * 2);
  __bf16* hdecb1 = (__bf16*)carve((size_t)BB * HH * 2);
  float*  hdecf0 = (float*)carve((size_t)BB * HH * 4);
  float*  hdecf1 = (float*)carve((size_t)BB * HH * 4);
  float*  cdec0  = (float*)carve((size_t)BB * HH * 4);
  float*  cdec1  = (float*)carve((size_t)BB * HH * 4);
  float*  hW     = (float*)carve((size_t)BB * AA * 4);
  float*  ctx    = (float*)carve((size_t)BB * HH * 4);

  // ---- weight prep ----
  auto tgrid = [](size_t n) { return dim3((unsigned)((n + 255) / 256)); };
  k_transpose_bf16<<<tgrid((size_t)FF * G4), 256, 0, stream>>>(Wih0, Wih0t, FF, G4);
  k_transpose_bf16<<<tgrid((size_t)HH * G4), 256, 0, stream>>>(Whh0, Whh0t, HH, G4);
  k_transpose_bf16<<<tgrid((size_t)HH * G4), 256, 0, stream>>>(Wih1, Wih1t, HH, G4);
  k_transpose_bf16<<<tgrid((size_t)HH * G4), 256, 0, stream>>>(Whh1, Whh1t, HH, G4);
  k_transpose_bf16<<<tgrid((size_t)HH * G4), 256, 0, stream>>>(WhhC, WhhCt, HH, G4);
  k_transpose_bf16<<<tgrid((size_t)2 * HH * AA), 256, 0, stream>>>(Wattn, WaT, 2 * HH, AA);
  k_bias_sum<<<8, 256, 0, stream>>>(bih0, bhh0, bsum0, G4);
  k_bias_sum<<<8, 256, 0, stream>>>(bih1, bhh1, bsum1, G4);
  k_bias_sum<<<8, 256, 0, stream>>>(bihC, bhhC, bsumC, G4);
  k_x_to_bf<<<tgrid((size_t)TT * BB * FF), 256, 0, stream>>>(x, xbf);

  const dim3 lstm_grid(HH / 16, BB / 16);  // (32, 8)

  // ---- encoder layer 0 ----
  hipMemsetAsync(henc0, 0, (size_t)BB * HH * 2, stream);
  hipMemsetAsync(cenc0, 0, (size_t)BB * HH * 4, stream);
  for (int t = 0; t < TT; ++t) {
    const __bf16* hp = (t & 1) ? henc1 : henc0;
    __bf16*       hn = (t & 1) ? henc0 : henc1;
    const float*  cp = (t & 1) ? cenc1 : cenc0;
    float*        cn = (t & 1) ? cenc0 : cenc1;
    k_lstm_step<<<lstm_grid, 128, 0, stream>>>(
        xbf + (size_t)t * BB * FF, FF, Wih0t, hp, Whh0t, bsum0, cp, cn, hn,
        out0bf + (size_t)t * BB * HH);
  }

  // ---- encoder layer 1 ----
  hipMemsetAsync(henc0, 0, (size_t)BB * HH * 2, stream);
  hipMemsetAsync(cenc0, 0, (size_t)BB * HH * 4, stream);
  for (int t = 0; t < TT; ++t) {
    const __bf16* hp = (t & 1) ? henc1 : henc0;
    __bf16*       hn = (t & 1) ? henc0 : henc1;
    const float*  cp = (t & 1) ? cenc1 : cenc0;
    float*        cn = (t & 1) ? cenc0 : cenc1;
    k_lstm_step<<<lstm_grid, 128, 0, stream>>>(
        out0bf + (size_t)t * BB * HH, HH, Wih1t, hp, Whh1t, bsum1, cp, cn, hn,
        lobf + (size_t)t * BB * HH);
  }
  // after 512 steps final (hT, cT) live in henc0 / cenc0

  // ---- encoder attention projection (once) ----
  k_encproj<<<(TT * BB / 16) * (AA / 16) / 8, 256, 0, stream>>>(lobf, WaT, battn, pebf);

  // ---- decoder ----
  for (int s = 0; s < HORZ; ++s) {
    const __bf16* hp = (s == 0) ? henc0 : ((s & 1) ? hdecb1 : hdecb0);
    const float*  cp = (s == 0) ? cenc0 : ((s & 1) ? cdec1 : cdec0);
    __bf16* hnb = (s & 1) ? hdecb0 : hdecb1;
    float*  hnf = (s & 1) ? hdecf0 : hdecf1;
    float*  cn  = (s & 1) ? cdec0 : cdec1;

    k_dechproj<<<32, 256, 0, stream>>>(hp, WaT + (size_t)HH * AA, hW);
    k_attn<<<BB, 256, 0, stream>>>(pebf, hW, vvec, lobf, ctx);
    k_dec_cell<<<lstm_grid, 128, 0, stream>>>(y, s, WihC, hp, WhhCt, bsumC,
                                              cp, cn, hnb, hnf);
    k_forecast<<<BB, 256, 0, stream>>>(hnf, ctx, Wf, bfp, out, s);
  }
}
